// LSTMNetBasic_730144440507
// MI455X (gfx1250) — matrix operand order repaired and checked by reading, not compiled, not run on hardware
//
#include <hip/hip_runtime.h>
#include <hip/hip_bf16.h>

// ---------------------------------------------------------------------------
// 2-layer LSTM (B=64, T=512, I=128, H=1024) + FC head for gfx1250 (MI455X).
// One fused kernel per time step: gates = h*Whh^T + x_t*Wih^T + bias, then
// elementwise update. v_wmma_f32_16x16x32_bf16, split-K over 8 waves with
// LDS reduction, SALU chunk control. Software pipeline enforced with
// __builtin_amdgcn_sched_barrier(0) so next-chunk loads are issued a full
// iteration ahead of their consuming WMMAs (the scheduler otherwise sinks
// loads to just before use, serializing L2 latency into the step).
// ---------------------------------------------------------------------------

typedef __attribute__((ext_vector_type(16))) __bf16 bf16x16;
typedef __attribute__((ext_vector_type(8)))  __bf16 bf16x8;
typedef __attribute__((ext_vector_type(8)))  float  f32x8;

#define LSTM_B   64
#define LSTM_T   512
#define LSTM_I   128
#define LSTM_H   1024
#define LSTM_4H  4096

// A operand (16x32 bf16): lane l holds row M=l%16, K = {kb..kb+15} contiguous
// (kb = kc + 16*(l>=16)). Two 16B loads, 16B apart.
__device__ inline bf16x16 load_a16x32(const __bf16* p) {
    bf16x8 lo = *reinterpret_cast<const bf16x8*>(p);
    bf16x8 hi = *reinterpret_cast<const bf16x8*>(p + 8);
    bf16x16 r;
#pragma unroll
    for (int i = 0; i < 8; ++i) { r[i] = lo[i]; r[8 + i] = hi[i]; }
    return r;
}

// B operand (32x16 bf16): lane l holds column N=l%16, K = {0..15} (l<16) or
// {16..31} (l>=16) — one contiguous 32B run per lane.
__device__ inline bf16x16 load_b32x16(const __bf16* p) {
    return *reinterpret_cast<const bf16x16*>(p);
}

__device__ inline f32x8 wmma_bf16(const bf16x16& a, const bf16x16& b, const f32x8& c) {
    return __builtin_amdgcn_wmma_f32_16x16x32_bf16(false, a, false, b,
                                                   (short)0, c, false, false);
}

// ---------------------------------------------------------------------------
// Fused LSTM step, templated on input width KI (128 layer0 / 1024 layer1).
// Grid: 256 blocks = (4 batch tiles x 64 h tiles); 256 threads = 8 waves
// splitting K; 32 KB LDS for the split-K reduction.
// ---------------------------------------------------------------------------
template <int KI>
__global__ __launch_bounds__(256, 2) void lstm_step_wmma(
    const __bf16* __restrict__ hprev,  // [64,1024] bf16
    __bf16* __restrict__ hnext,        // [64,1024] bf16
    float* __restrict__ c_state,       // [64,1024] f32 (exclusively owned)
    const __bf16* __restrict__ xin,    // row b at xin + b*(T*KI), length KI
    const __bf16* __restrict__ Whh,    // [4096,1024] bf16 row-major
    const __bf16* __restrict__ Wih,    // [4096,KI]   bf16 row-major
    const float* __restrict__ bias,    // [4096] = b_ih + b_hh
    __bf16* __restrict__ hseq)         // row b at hseq + b*(T*H)
{
    __shared__ float smem[8 * 4 * 8 * 32];  // [wave][gate][vgpr][lane] = 32 KB

    const int tid  = threadIdx.x;
    // Wave index in an SGPR: all chunk control is SALU; EXEC untouched.
    const int wave = __builtin_amdgcn_readfirstlane(tid >> 5);
    const int lane = tid & 31;
    const int m    = blockIdx.x >> 6;   // batch tile 0..3
    const int n    = blockIdx.x & 63;   // h tile    0..63

    const int r16   = lane & 15;        // A: row in tile; B: column in tile
    const int koffA = (lane >> 4) << 4; // 0 or 16
    const int koffB = (lane >> 4) << 4; // 0 or 16

    // Per-lane base pointers (chunk K offset added per iteration).
    const __bf16* aH = hprev + (size_t)(m * 16 + r16) * LSTM_H + koffA;
    const __bf16* bH = Whh   + (size_t)(n * 16 + r16) * LSTM_H + koffB;
    const __bf16* aX = xin   + (size_t)(m * 16 + r16) * (LSTM_T * KI) + koffA;
    const __bf16* bX = Wih   + (size_t)(n * 16 + r16) * KI + koffB;

    f32x8 acc[4];
#pragma unroll
    for (int g = 0; g < 4; ++g)
#pragma unroll
        for (int i = 0; i < 8; ++i) acc[g][i] = 0.0f;

    // Chunk sequence: q = 0..3 -> h part (chunk = wave + 8q), q = 4..7 -> x
    // part when KI == 1024. q is compile-time under full unroll.
    auto loadA = [&](int q) -> bf16x16 {
        if (q < 4) return load_a16x32(aH + ((wave + 8 * q) << 5));
        return load_a16x32(aX + ((wave + 8 * (q - 4)) << 5));
    };
    auto loadB = [&](int q, int g) -> bf16x16 {
        if (q < 4)
            return load_b32x16(bH + (size_t)g * LSTM_H * LSTM_H
                                  + ((wave + 8 * q) << 5));
        return load_b32x16(bX + (size_t)g * LSTM_H * KI
                              + ((wave + 8 * (q - 4)) << 5));
    };

    constexpr int NQ = (KI == 1024) ? 8 : 4;   // pipelined iterations per wave

    bf16x16 curA  = loadA(0);
    bf16x16 curB0 = loadB(0, 0), curB1 = loadB(0, 1),
            curB2 = loadB(0, 2), curB3 = loadB(0, 3);

#pragma unroll
    for (int q = 0; q < NQ; ++q) {
        const bool more = (q + 1 < NQ);
        bf16x16 nA, nB0, nB1, nB2, nB3;
        if (more) {                       // prefetch chunk q+1 (10 b128 loads)
            nA  = loadA(q + 1);
            nB0 = loadB(q + 1, 0);
            nB1 = loadB(q + 1, 1);
            nB2 = loadB(q + 1, 2);
            nB3 = loadB(q + 1, 3);
        }
        // Fence the schedule: prefetch loads may not sink below this point,
        // WMMAs may not hoist above it. Keeps chunk q+1 in flight during the
        // 4 WMMAs of chunk q -> partial s_wait_loadcnt instead of 0.
        __builtin_amdgcn_sched_barrier(0);
        acc[0] = wmma_bf16(curA, curB0, acc[0]);
        acc[1] = wmma_bf16(curA, curB1, acc[1]);
        acc[2] = wmma_bf16(curA, curB2, acc[2]);
        acc[3] = wmma_bf16(curA, curB3, acc[3]);
        if (more) {
            curA = nA; curB0 = nB0; curB1 = nB1; curB2 = nB2; curB3 = nB3;
        }
    }

    // KI = 128 tail: 4 x-chunks total, one each for waves 0-3 (SGPR branch).
    if constexpr (KI == 128) {
        if (wave < 4) {
            bf16x16 ax = loadA(4);
            bf16x16 x0 = loadB(4, 0), x1 = loadB(4, 1),
                    x2 = loadB(4, 2), x3 = loadB(4, 3);
            acc[0] = wmma_bf16(ax, x0, acc[0]);
            acc[1] = wmma_bf16(ax, x1, acc[1]);
            acc[2] = wmma_bf16(ax, x2, acc[2]);
            acc[3] = wmma_bf16(ax, x3, acc[3]);
        }
    }

    // ---- split-K reduction via LDS ----
#pragma unroll
    for (int g = 0; g < 4; ++g)
#pragma unroll
        for (int j = 0; j < 8; ++j)
            smem[(((wave * 4 + g) * 8 + j) << 5) + lane] = acc[g][j];

    __syncthreads();

    // Elementwise phase: thread tid owns tile element (row r, col cc).
    // WMMA C layout: VGPR j holds (M=j, lanes 0-15) and (M=8+j, lanes 16-31).
    const int r  = tid >> 4;                  // 0..15
    const int cc = tid & 15;                  // 0..15
    const int l2 = cc + ((r >= 8) ? 16 : 0);
    const int j2 = r & 7;

    float gsum[4];
#pragma unroll
    for (int g = 0; g < 4; ++g) {
        float s = 0.0f;
#pragma unroll
        for (int w = 0; w < 8; ++w)
            s += smem[(((w * 4 + g) * 8 + j2) << 5) + l2];
        gsum[g] = s;
    }

    const int bidx = m * 16 + r;
    const int hcol = n * 16 + cc;

    const float ig = 1.0f / (1.0f + __expf(-(gsum[0] + bias[0 * LSTM_H + hcol])));
    const float fg = 1.0f / (1.0f + __expf(-(gsum[1] + bias[1 * LSTM_H + hcol])));
    const float gg = tanhf(gsum[2] + bias[2 * LSTM_H + hcol]);
    const float og = 1.0f / (1.0f + __expf(-(gsum[3] + bias[3 * LSTM_H + hcol])));

    const size_t ci = (size_t)bidx * LSTM_H + hcol;
    const float cn  = fg * c_state[ci] + ig * gg;
    c_state[ci]     = cn;
    const float hn  = og * tanhf(cn);
    const __bf16 hb = (__bf16)hn;
    hnext[ci] = hb;
    hseq[(size_t)bidx * (LSTM_T * LSTM_H) + hcol] = hb;
}

// ---------------------------------------------------------------------------
// Small helper kernels
// ---------------------------------------------------------------------------
__global__ void f32_to_bf16_kernel(const float* __restrict__ src,
                                   __bf16* __restrict__ dst, int n) {
    int i = blockIdx.x * blockDim.x + threadIdx.x;
    if (i < n) dst[i] = (__bf16)src[i];
}

__global__ void copy_f32_kernel(const float* __restrict__ src,
                                float* __restrict__ dst, int n) {
    int i = blockIdx.x * blockDim.x + threadIdx.x;
    if (i < n) dst[i] = src[i];
}

__global__ void bias_fold_kernel(const float* __restrict__ a,
                                 const float* __restrict__ b,
                                 float* __restrict__ out, int n) {
    int i = blockIdx.x * blockDim.x + threadIdx.x;
    if (i < n) out[i] = a[i] + b[i];
}

// FC head: out[bt] = dot(h2[bt,:], W_fc) + b_fc. One wave per output.
__global__ __launch_bounds__(256) void fc_head_kernel(
    const __bf16* __restrict__ h2, const float* __restrict__ Wfc,
    const float* __restrict__ bfc, float* __restrict__ out) {
    const int wid  = blockIdx.x * 8 + (threadIdx.x >> 5);   // 0..32767 == b*T+t
    const int lane = threadIdx.x & 31;
    const __bf16* row = h2 + (size_t)wid * LSTM_H;
    float s = 0.0f;
    for (int k = lane; k < LSTM_H; k += 32) s += (float)row[k] * Wfc[k];
#pragma unroll
    for (int off = 16; off; off >>= 1) s += __shfl_xor(s, off, 32);
    if (lane == 0) out[wid] = s + bfc[0];
}

// ---------------------------------------------------------------------------
// Host launcher
// ---------------------------------------------------------------------------
extern "C" void kernel_launch(void* const* d_in, const int* in_sizes, int n_in,
                              void* d_out, int out_size, void* d_ws, size_t ws_size,
                              hipStream_t stream) {
    (void)in_sizes; (void)n_in; (void)out_size; (void)ws_size;

    const float* x     = (const float*)d_in[0];   // [64,512,128]
    const float* h0    = (const float*)d_in[1];   // [2,64,1024]
    const float* c0    = (const float*)d_in[2];   // [2,64,1024]
    const float* Wih0  = (const float*)d_in[3];   // [4096,128]
    const float* Whh0  = (const float*)d_in[4];   // [4096,1024]
    const float* bih0  = (const float*)d_in[5];
    const float* bhh0  = (const float*)d_in[6];
    const float* Wih1  = (const float*)d_in[7];   // [4096,1024]
    const float* Whh1  = (const float*)d_in[8];   // [4096,1024]
    const float* bih1  = (const float*)d_in[9];
    const float* bhh1  = (const float*)d_in[10];
    const float* Wfc   = (const float*)d_in[11];  // [1,1024]
    const float* bfc   = (const float*)d_in[12];  // [1]
    float* out = (float*)d_out;                   // [64,512]

    const int BT = LSTM_B * LSTM_T;               // 32768
    const int BH = LSTM_B * LSTM_H;               // 65536

    // ---- workspace carve-up (all offsets 256B-aligned) ----
    char*  ws  = (char*)d_ws;
    size_t off = 0;
    auto carve = [&](size_t bytes) {
        void* p = ws + off;
        off += (bytes + 255) & ~(size_t)255;
        return p;
    };
    __bf16* x_bf    = (__bf16*)carve((size_t)BT * LSTM_I * 2);         // 8 MB
    __bf16* Wih0_bf = (__bf16*)carve((size_t)LSTM_4H * LSTM_I * 2);    // 1 MB
    __bf16* Whh0_bf = (__bf16*)carve((size_t)LSTM_4H * LSTM_H * 2);    // 8 MB
    __bf16* Wih1_bf = (__bf16*)carve((size_t)LSTM_4H * LSTM_H * 2);    // 8 MB
    __bf16* Whh1_bf = (__bf16*)carve((size_t)LSTM_4H * LSTM_H * 2);    // 8 MB
    float*  bsum0   = (float*) carve((size_t)LSTM_4H * 4);
    float*  bsum1   = (float*) carve((size_t)LSTM_4H * 4);
    __bf16* hA      = (__bf16*)carve((size_t)BH * 2);
    __bf16* hB      = (__bf16*)carve((size_t)BH * 2);
    float*  c_st    = (float*) carve((size_t)BH * 4);
    __bf16* h1_seq  = (__bf16*)carve((size_t)BT * LSTM_H * 2);         // 64 MB
    __bf16* h2_seq  = (__bf16*)carve((size_t)BT * LSTM_H * 2);         // 64 MB

    // ---- conversions (re-run every call: deterministic) ----
    auto conv = [&](const float* s, __bf16* d, int n) {
        f32_to_bf16_kernel<<<(n + 255) / 256, 256, 0, stream>>>(s, d, n);
    };
    conv(x,    x_bf,    BT * LSTM_I);
    conv(Wih0, Wih0_bf, LSTM_4H * LSTM_I);
    conv(Whh0, Whh0_bf, LSTM_4H * LSTM_H);
    conv(Wih1, Wih1_bf, LSTM_4H * LSTM_H);
    conv(Whh1, Whh1_bf, LSTM_4H * LSTM_H);
    bias_fold_kernel<<<(LSTM_4H + 255) / 256, 256, 0, stream>>>(bih0, bhh0, bsum0, LSTM_4H);
    bias_fold_kernel<<<(LSTM_4H + 255) / 256, 256, 0, stream>>>(bih1, bhh1, bsum1, LSTM_4H);

    // ---- layer 0 ----
    conv(h0, hA, BH);
    copy_f32_kernel<<<(BH + 255) / 256, 256, 0, stream>>>(c0, c_st, BH);
    for (int t = 0; t < LSTM_T; ++t) {
        __bf16* hp = (t & 1) ? hB : hA;
        __bf16* hn = (t & 1) ? hA : hB;
        lstm_step_wmma<LSTM_I><<<256, 256, 0, stream>>>(
            hp, hn, c_st, x_bf + (size_t)t * LSTM_I,
            Whh0_bf, Wih0_bf, bsum0, h1_seq + (size_t)t * LSTM_H);
    }

    // ---- layer 1 (input = h1 sequence) ----
    conv(h0 + BH, hA, BH);
    copy_f32_kernel<<<(BH + 255) / 256, 256, 0, stream>>>(c0 + BH, c_st, BH);
    for (int t = 0; t < LSTM_T; ++t) {
        __bf16* hp = (t & 1) ? hB : hA;
        __bf16* hn = (t & 1) ? hA : hB;
        lstm_step_wmma<LSTM_H><<<256, 256, 0, stream>>>(
            hp, hn, c_st, h1_seq + (size_t)t * LSTM_H,
            Whh1_bf, Wih1_bf, bsum1, h2_seq + (size_t)t * LSTM_H);
    }

    // ---- FC head ----
    fc_head_kernel<<<BT / 8, 256, 0, stream>>>(h2_seq, Wfc, bfc, out);
}
